// nGCNN_970662609200
// MI455X (gfx1250) — matrix operand order repaired and checked by reading, not compiled, not run on hardware
//
#include <hip/hip_runtime.h>
#include <hip/hip_bf16.h>

// ---------------------------------------------------------------------------
// GraphConv GNN for MI455X (gfx1250): bf16 WMMA GEMMs + atomic edge scatter.
// Activations staged as K-padded bf16 so every WMMA fragment load is an
// unconditional vectorized global_load_b128 (no lane-varying guards).
// ---------------------------------------------------------------------------

typedef __attribute__((ext_vector_type(16))) __bf16 bf16x16;
typedef __attribute__((ext_vector_type(8)))  __bf16 bf16x8;
typedef __attribute__((ext_vector_type(8)))  float  v8f;

#define NNODES 50000

// ------------------------- utility kernels ---------------------------------

__global__ void zero_kernel(float* p, long long n) {
    long long i = (long long)blockIdx.x * blockDim.x + threadIdx.x;
    long long stride = (long long)gridDim.x * blockDim.x;
    for (; i < n; i += stride) p[i] = 0.0f;
}

// (rows, K) f32  ->  (rows, Kp) bf16, zero-padded columns K..Kp-1
__global__ void cvt_pad_bf16_kernel(const float* __restrict__ src,
                                    __bf16* __restrict__ dst,
                                    int rows, int K, int Kp) {
    long long total = (long long)rows * Kp;
    long long idx = (long long)blockIdx.x * blockDim.x + threadIdx.x;
    long long stride = (long long)gridDim.x * blockDim.x;
    for (; idx < total; idx += stride) {
        int r = (int)(idx / Kp);
        int c = (int)(idx - (long long)r * Kp);
        float v = (c < K) ? src[(long long)r * K + c] : 0.0f;
        dst[idx] = (__bf16)v;
    }
}

// ----------------------- edge scatter (segment_sum) ------------------------
// One wave per edge; 32 lanes stride the feature dim -> coalesced 128B
// transactions; accumulation via global_atomic_add_f32.

__global__ void scatter_kernel(const float* __restrict__ X,
                               const int* __restrict__ src,
                               const int* __restrict__ dst,
                               const float* __restrict__ w,
                               float* __restrict__ Agg, int E, int F) {
    int wave  = blockIdx.x * (blockDim.x >> 5) + (threadIdx.x >> 5);
    int lane  = threadIdx.x & 31;
    int nwave = gridDim.x * (blockDim.x >> 5);
    for (int e = wave; e < E; e += nwave) {
        int   s  = src[e];
        int   d  = dst[e];
        float we = w[e];
        const float* xp = X   + (long long)s * F;
        float*       ap = Agg + (long long)d * F;
        for (int f = lane; f < F; f += 32)
            atomicAdd(ap + f, we * xp[f]);
    }
}

// --------------------------- WMMA dual GEMM --------------------------------
// Out[M,Nout] = Abf[M,K] @ Wrel^T + Xbf[M,K] @ Wroot^T + bias (+opt. relu)
// All operands bf16 with row stride K (padded, multiple of 32).
// One wave computes a 16x32 output tile (two 16x16 WMMA accumulators);
// the A fragment is reused across both B columns.

__device__ __forceinline__ bf16x16 load_a_frag(const __bf16* __restrict__ row,
                                               int kb, int h) {
    // A 16x32 layout: lane r+16h, elements 0..7 -> K = kb+16h+0..7,
    //                             elements 8..15 -> K = kb+16h+8+0..7
    bf16x8 lo = *(const bf16x8*)(row + kb + 16 * h);
    bf16x8 hi = *(const bf16x8*)(row + kb + 16 * h + 8);
    bf16x16 a;
#pragma unroll
    for (int i = 0; i < 8; ++i) { a[i] = lo[i]; a[i + 8] = hi[i]; }
    return a;
}

#define WMMA_BF16(A, B, C) \
    __builtin_amdgcn_wmma_f32_16x16x32_bf16(false, (A), false, (B), (short)0, \
                                            (C), false, false)

template <int K>
__global__ void gemm_dual_kernel(const __bf16* __restrict__ Abf,
                                 const __bf16* __restrict__ Xbf,
                                 const __bf16* __restrict__ Wrel,
                                 const __bf16* __restrict__ Wroot,
                                 const float* __restrict__ bias,
                                 float* __restrict__ Out,
                                 __bf16* __restrict__ Obf,   // may be null
                                 int M, int Nout, int relu) {
    int wave = blockIdx.x * (blockDim.x >> 5) + (threadIdx.x >> 5);
    int lane = threadIdx.x & 31;
    int ntiles = Nout >> 5;                  // 32-wide output tiles
    int mt = wave / ntiles;
    int nt = wave - mt * ntiles;
    if (mt >= (M >> 4)) return;              // wave-uniform exit: EXEC all-1s

    int m0 = mt << 4, n0 = nt << 5;
    int r = lane & 15;
    int h = lane >> 4;

    const __bf16* arow = Abf + (long long)(m0 + r) * K;
    const __bf16* xrow = Xbf + (long long)(m0 + r) * K;
    // B 32x16 layout: lane n+16h holds col n, 16 contiguous K at kb+16h.
    const __bf16* wr0 = Wrel  + (long long)(n0 + r) * K      + 16 * h;
    const __bf16* wr1 = Wrel  + (long long)(n0 + 16 + r) * K + 16 * h;
    const __bf16* wo0 = Wroot + (long long)(n0 + r) * K      + 16 * h;
    const __bf16* wo1 = Wroot + (long long)(n0 + 16 + r) * K + 16 * h;

    v8f acc0, acc1;
#pragma unroll
    for (int i = 0; i < 8; ++i) { acc0[i] = 0.0f; acc1[i] = 0.0f; }

#pragma unroll 2
    for (int kb = 0; kb < K; kb += 32) {
        bf16x16 a1 = load_a_frag(arow, kb, h);
        bf16x16 b0 = *(const bf16x16*)(wr0 + kb);
        bf16x16 b1 = *(const bf16x16*)(wr1 + kb);
        acc0 = WMMA_BF16(a1, b0, acc0);
        acc1 = WMMA_BF16(a1, b1, acc1);

        bf16x16 a2 = load_a_frag(xrow, kb, h);
        bf16x16 c0 = *(const bf16x16*)(wo0 + kb);
        bf16x16 c1 = *(const bf16x16*)(wo1 + kb);
        acc0 = WMMA_BF16(a2, c0, acc0);
        acc1 = WMMA_BF16(a2, c1, acc1);
    }

    // C/D layout: lane n+16h, VGPR v -> row (v + 8h), col n
    int col0 = n0 + r;
    int col1 = col0 + 16;
    float bv0 = bias[col0];
    float bv1 = bias[col1];
#pragma unroll
    for (int v = 0; v < 8; ++v) {
        long long orow = m0 + v + 8 * h;
        float f0 = acc0[v] + bv0;
        float f1 = acc1[v] + bv1;
        if (relu) { f0 = fmaxf(f0, 0.0f); f1 = fmaxf(f1, 0.0f); }
        Out[orow * Nout + col0] = f0;
        Out[orow * Nout + col1] = f1;
        if (Obf) {                            // bf16 copy for next layer's GEMM
            Obf[orow * Nout + col0] = (__bf16)f0;
            Obf[orow * Nout + col1] = (__bf16)f1;
        }
    }
}

// ------------------------------ GraphNorm ----------------------------------

__global__ void colstat_kernel(const float* __restrict__ H,
                               float* __restrict__ colsum,
                               float* __restrict__ colsumsq,
                               int Nrows, int F) {
    int c = threadIdx.x;                      // blockDim.x == F
    float s = 0.0f, s2 = 0.0f;
    for (int r = blockIdx.x; r < Nrows; r += gridDim.x) {
        float v = H[(long long)r * F + c];
        s += v; s2 += v * v;
    }
    atomicAdd(colsum + c, s);
    atomicAdd(colsumsq + c, s2);
}

__global__ void gnorm_kernel(const float* __restrict__ H,
                             const float* __restrict__ colsum,
                             const float* __restrict__ colsumsq,
                             const float* __restrict__ w,
                             const float* __restrict__ b,
                             const float* __restrict__ ms,
                             float* __restrict__ Outf,
                             __bf16* __restrict__ Outbf,
                             int Nrows, int F) {
    long long total = (long long)Nrows * F;
    long long idx = (long long)blockIdx.x * blockDim.x + threadIdx.x;
    long long stride = (long long)gridDim.x * blockDim.x;
    float invN = 1.0f / (float)Nrows;
    for (; idx < total; idx += stride) {
        int c = (int)(idx % F);
        float mu  = colsum[c] * invN;
        float ex2 = colsumsq[c] * invN;
        float m   = ms[c];
        // var(x - m*mu) = E[x^2] - 2*m*mu^2 + m^2*mu^2
        float var = ex2 - 2.0f * m * mu * mu + m * m * mu * mu;
        float inv = rsqrtf(var + 1e-5f);
        float v = (H[idx] - m * mu) * inv * w[c] + b[c];
        v = fmaxf(v, 0.0f);                   // relu(graph_norm(h))
        Outf[idx]  = v;
        Outbf[idx] = (__bf16)v;
    }
}

// ------------------------------ launcher -----------------------------------

extern "C" void kernel_launch(void* const* d_in, const int* in_sizes, int n_in,
                              void* d_out, int out_size, void* d_ws, size_t ws_size,
                              hipStream_t stream) {
    const int N = NNODES;
    const int E = in_sizes[1] / 2;

    const float* x        = (const float*)d_in[0];
    const int*   src      = (const int*)d_in[1];
    const int*   dst      = (const int*)d_in[1] + E;
    const float* eattr    = (const float*)d_in[2];
    const float* Wrel[4]  = {(const float*)d_in[3], (const float*)d_in[6],
                             (const float*)d_in[9], (const float*)d_in[12]};
    const float* brel[4]  = {(const float*)d_in[4], (const float*)d_in[7],
                             (const float*)d_in[10], (const float*)d_in[13]};
    const float* Wroot[4] = {(const float*)d_in[5], (const float*)d_in[8],
                             (const float*)d_in[11], (const float*)d_in[14]};
    const float* gn_w  = (const float*)d_in[15];
    const float* gn_b  = (const float*)d_in[16];
    const float* gn_ms = (const float*)d_in[17];

    const int Fin[4]  = {144, 256, 512, 256};
    const int Kp[4]   = {160, 256, 512, 256};   // K padded to multiple of 32
    const int Fout[4] = {256, 512, 256, 128};

    // ---- workspace layout ----
    char* ws = (char*)d_ws;
    size_t off = 0;
    auto carve = [&](size_t bytes) {
        void* p = ws + off;
        off += (bytes + 255) & ~(size_t)255;
        return p;
    };
    float*  agg   = (float*)carve((size_t)N * 512 * 4);   // f32 aggregation
    __bf16* aggbf = (__bf16*)carve((size_t)N * 512 * 2);  // padded bf16 copy
    __bf16* xbf   = (__bf16*)carve((size_t)N * 160 * 2);  // padded input x
    float*  h1f   = (float*)carve((size_t)N * 256 * 4);
    __bf16* h1bf  = (__bf16*)carve((size_t)N * 256 * 2);
    float*  h2f   = (float*)carve((size_t)N * 512 * 4);
    __bf16* h2bf  = (__bf16*)carve((size_t)N * 512 * 2);
    __bf16* wbf[8];
    for (int l = 0; l < 4; ++l) {
        wbf[2 * l]     = (__bf16*)carve((size_t)Fout[l] * Kp[l] * 2); // rel
        wbf[2 * l + 1] = (__bf16*)carve((size_t)Fout[l] * Kp[l] * 2); // root
    }
    float* colsum   = (float*)carve(256 * 4);
    float* colsumsq = (float*)carve(256 * 4);
    (void)ws_size; (void)n_in; (void)out_size;

    // buffer aliases for dead-buffer reuse
    float*  h3f  = h1f;    // h1f dead after layer-2 scatter
    float*  hnf  = h2f;    // h2f dead after layer-3 scatter
    __bf16* hnbf = h1bf;   // h1bf dead after layer-2 GEMM

    // ---- one-time conversions (weights + input x), padded ----
    for (int l = 0; l < 4; ++l) {
        int n = Fout[l] * Kp[l];
        int blocks = (n + 255) / 256;
        cvt_pad_bf16_kernel<<<blocks, 256, 0, stream>>>(Wrel[l],  wbf[2 * l],
                                                        Fout[l], Fin[l], Kp[l]);
        cvt_pad_bf16_kernel<<<blocks, 256, 0, stream>>>(Wroot[l], wbf[2 * l + 1],
                                                        Fout[l], Fin[l], Kp[l]);
    }
    cvt_pad_bf16_kernel<<<4096, 256, 0, stream>>>(x, xbf, N, 144, 160);

    const float*  in_f[4]  = {x,    h1f,  h2f,  hnf};
    const __bf16* in_bf[4] = {xbf,  h1bf, h2bf, hnbf};
    float*  out_f[4]  = {h1f,  h2f,  h3f, (float*)d_out};
    __bf16* out_bf[4] = {h1bf, h2bf, nullptr, nullptr};  // h3 goes via gnorm
    int relu_l[4] = {1, 1, 0, 0};

    for (int l = 0; l < 4; ++l) {
        int F = Fin[l], Fo = Fout[l], KP = Kp[l];
        // 1. zero + edge scatter into f32 agg
        zero_kernel<<<2048, 256, 0, stream>>>(agg, (long long)N * F);
        scatter_kernel<<<4096, 256, 0, stream>>>(in_f[l], src, dst, eattr,
                                                 agg, E, F);
        // 2. agg f32 -> padded bf16
        cvt_pad_bf16_kernel<<<4096, 256, 0, stream>>>(agg, aggbf, N, F, KP);
        // 3. dual WMMA GEMM (16x32 tile per wave, 8 waves per block)
        int tiles  = (N >> 4) * (Fo >> 5);
        int blocks = (tiles + 7) / 8;
        if (KP == 160)
            gemm_dual_kernel<160><<<blocks, 256, 0, stream>>>(
                aggbf, in_bf[l], wbf[2 * l], wbf[2 * l + 1], brel[l],
                out_f[l], out_bf[l], N, Fo, relu_l[l]);
        else if (KP == 256)
            gemm_dual_kernel<256><<<blocks, 256, 0, stream>>>(
                aggbf, in_bf[l], wbf[2 * l], wbf[2 * l + 1], brel[l],
                out_f[l], out_bf[l], N, Fo, relu_l[l]);
        else
            gemm_dual_kernel<512><<<blocks, 256, 0, stream>>>(
                aggbf, in_bf[l], wbf[2 * l], wbf[2 * l + 1], brel[l],
                out_f[l], out_bf[l], N, Fo, relu_l[l]);

        if (l == 2) {
            // GraphNorm + ReLU: h3f -> hnf (f32) + hnbf (bf16)
            zero_kernel<<<1, 512, 0, stream>>>(colsum, 512); // sum + sumsq
            colstat_kernel<<<512, 256, 0, stream>>>(h3f, colsum, colsumsq,
                                                    N, 256);
            gnorm_kernel<<<4096, 256, 0, stream>>>(h3f, colsum, colsumsq,
                                                   gn_w, gn_b, gn_ms,
                                                   hnf, hnbf, N, 256);
        }
    }
}
